// SeqModel_57793079935458
// MI455X (gfx1250) — matrix operand order repaired and checked by reading, not compiled, not run on hardware
//
#include <hip/hip_runtime.h>

// ---------------------------------------------------------------------------
// LSTM scan (H=64, IN=1, T=524288) + linear head, single persistent workgroup.
//
// Latency-bound sequential recurrence: one block, 128 threads = 4 wave32s
// (one per SIMD of a WGP). Wave w computes gate w's 64 pre-activations per
// step with v_wmma_f32_16x16x32_f16 using the "replicated-A" matvec trick:
//   A(16x32)  = h chunk replicated across all 16 rows   (4 ds_load_b128/step)
//   B(32x16)  = W_hh^T tile, static in VGPRs for all T steps
//   D(16x16)  = every row identical => each lane holds its tile's outputs
// Threads 0..63 keep (c_j, h_j) in registers and run the elementwise update
// with v_exp/v_rcp transcendentals. Two s_barrier's per step.
// ---------------------------------------------------------------------------

typedef __attribute__((ext_vector_type(16))) _Float16 v16h;
typedef __attribute__((ext_vector_type(8)))  _Float16 v8h;
typedef __attribute__((ext_vector_type(8)))  float    v8f;

#define HH        64
#define TT        524288
#define NGATES    256
#define SEQ_TILE  128

__device__ __forceinline__ float fast_sigmoid(float x) {
  // 1 / (1 + 2^(-x*log2(e)))  -> v_exp_f32 + v_rcp_f32
  float e = __builtin_amdgcn_exp2f(-1.4426950408889634f * x);
  return __builtin_amdgcn_rcpf(1.0f + e);
}
__device__ __forceinline__ float fast_tanh(float x) {
  // 2 / (1 + 2^(-2x*log2(e))) - 1
  float e = __builtin_amdgcn_exp2f(-2.8853900817779268f * x);
  return __builtin_amdgcn_rcpf(1.0f + e) * 2.0f - 1.0f;
}

__global__ void __launch_bounds__(128, 1)
lstm_scan_kernel(const float* __restrict__ seq,
                 const float* __restrict__ W_ih,
                 const float* __restrict__ W_hh,
                 const float* __restrict__ b_ih,
                 const float* __restrict__ b_hh,
                 const float* __restrict__ W_lin,
                 const float* __restrict__ b_lin,
                 float* __restrict__ out)
{
  __shared__ _Float16 shH[HH];         // h_t (f16) -> WMMA A operand source
  __shared__ float    shG[NGATES];     // gate pre-activations (i,f,g,o)
  __shared__ float    shX[SEQ_TILE];   // staged seq values
  __shared__ float    shR[HH];         // head partial products

  const int tid   = threadIdx.x;
  const int lane  = tid & 31;
  const int wave  = tid >> 5;          // wave w == gate w (PyTorch order i,f,g,o)
  const int lrow  = lane & 15;         // row within a 16-wide output tile
  const int khalf = (lane < 16) ? 0 : 1;

  // ---- Preload W_hh^T tiles into the 16-bit WMMA B layout (VGPR-resident).
  // B(32x16) f16, contiguous K order chosen here: VGPR v, lanes 0-15: N=lane, K={2v,2v+1};
  // lanes 16-31: N=lane-16, K={16+2v,17+2v}. So per lane the 16 vector
  // elements are 16 *consecutive* K values of row N -> contiguous loads.
  v16h B[4][2];
  float wih[4], bsum[4];
  #pragma unroll
  for (int t = 0; t < 4; ++t) {
    const int row = wave * 64 + t * 16 + lrow;       // gate pre-act index
    wih[t]  = W_ih[row];                             // IN == 1
    bsum[t] = b_ih[row] + b_hh[row];
    #pragma unroll
    for (int c = 0; c < 2; ++c) {
      const float* src = W_hh + row * HH + c * 32 + khalf * 16;
      #pragma unroll
      for (int e = 0; e < 16; ++e)
        B[t][c][e] = (_Float16)src[e];
    }
  }

  // Per-thread recurrent state: thread j (<64) owns c[j], h[j].
  float cst = 0.0f, hst = 0.0f;
  const float wlin = (tid < HH) ? W_lin[tid] : 0.0f;

  if (tid < HH) shH[tid] = (_Float16)0.0f;           // h_0 = 0
  __syncthreads();

  // A-operand 16-bit layout splits K by lane-half only (matches B):
  // chunk c: elements 0-7  = h[32c + 16*khalf + 0..7]
  //          elements 8-15 = h[32c + 16*khalf + 8 + 0..7]
  const int sel8 = khalf * 8;

  for (int t = 0; t < TT; ++t) {
    // ---- stage 128 input samples into LDS, prefetch well ahead ----
    if ((t & (SEQ_TILE - 1)) == 0) {
      shX[tid] = seq[t + tid];                        // TT % 128 == 0
      int pf = t + 4 * SEQ_TILE + tid;
      __builtin_prefetch(seq + (pf < TT ? pf : TT - 1), 0, 1);
      __syncthreads();
    }
    const float x = shX[t & (SEQ_TILE - 1)];

    // ---- build replicated-h A operands from LDS (4x ds_load_b128) ----
    union { v16h v; v8h p[2]; } A0, A1;
    A0.p[0] = *(const v8h*)(shH + 0  + 2 * sel8);
    A0.p[1] = *(const v8h*)(shH + 8  + 2 * sel8);
    A1.p[0] = *(const v8h*)(shH + 32 + 2 * sel8);
    A1.p[1] = *(const v8h*)(shH + 40 + 2 * sel8);

    // ---- 4 tiles x (2 chained WMMAs over K=64); D rows all identical ----
    float gpre[4];
    #pragma unroll
    for (int tt = 0; tt < 4; ++tt) {
      v8f acc = {};
      acc = __builtin_amdgcn_wmma_f32_16x16x32_f16(
              false, A0.v, false, B[tt][0], (short)0, acc, false, false);
      acc = __builtin_amdgcn_wmma_f32_16x16x32_f16(
              false, A1.v, false, B[tt][1], (short)0, acc, false, false);
      gpre[tt] = acc[0] + (x * wih[tt] + bsum[tt]);   // + input & bias terms
    }
    if (lane < 16) {
      #pragma unroll
      for (int tt = 0; tt < 4; ++tt)
        shG[wave * 64 + tt * 16 + lrow] = gpre[tt];
    }
    __syncthreads();

    // ---- elementwise recurrence: thread j owns state element j ----
    if (tid < HH) {
      float gi = fast_sigmoid(shG[tid]);
      float gf = fast_sigmoid(shG[HH + tid]);
      float gg = fast_tanh   (shG[2 * HH + tid]);
      float go = fast_sigmoid(shG[3 * HH + tid]);
      cst = gf * cst + gi * gg;
      hst = go * fast_tanh(cst);
      shH[tid] = (_Float16)hst;
    }
    __syncthreads();
  }

  // ---- linear head on h_T: out = h_T . W_lin + b_lin ----
  if (tid < HH) shR[tid] = hst * wlin;
  __syncthreads();
  if (tid == 0) {
    float s = b_lin[0];
    #pragma unroll
    for (int j = 0; j < HH; ++j) s += shR[j];
    out[0] = s;
  }
}

extern "C" void kernel_launch(void* const* d_in, const int* in_sizes, int n_in,
                              void* d_out, int out_size, void* d_ws, size_t ws_size,
                              hipStream_t stream) {
  (void)in_sizes; (void)n_in; (void)out_size; (void)d_ws; (void)ws_size;
  const float* seq   = (const float*)d_in[0];
  const float* W_ih  = (const float*)d_in[1];
  const float* W_hh  = (const float*)d_in[2];
  const float* b_ih  = (const float*)d_in[3];
  const float* b_hh  = (const float*)d_in[4];
  const float* W_lin = (const float*)d_in[5];
  const float* b_lin = (const float*)d_in[6];
  float* out = (float*)d_out;

  // One persistent workgroup: 4 wave32s on one WGP carry the whole scan.
  lstm_scan_kernel<<<1, 128, 0, stream>>>(seq, W_ih, W_hh, b_ih, b_hh,
                                          W_lin, b_lin, out);
}
